// VarianceAdaptor_37452114821288
// MI455X (gfx1250) — matrix operand order repaired and checked by reading, not compiled, not run on hardware
//
#include <hip/hip_runtime.h>
#include <hip/hip_bf16.h>
#include <stdint.h>

// Problem constants (match reference)
#define B_   32
#define S_   512
#define H_   256
#define F_   256
#define T_   2048
#define KW   3
#define KDIM (KW * F_)        // 768
#define NCHUNK (KDIM / 32)    // 24 WMMA K-steps
#define LDA  (KDIM + 8)       // padded LDS row stride (ushorts) to break bank conflicts
#define SETSZ (16 * NCHUNK * 32 * 16)  // packed bf16 elements per conv-weight set = 196608

typedef __attribute__((ext_vector_type(16))) __bf16 v16bf;
typedef __attribute__((ext_vector_type(8)))  float  v8f;

__device__ __forceinline__ unsigned short f2bf_raw(float f) {
  union { float f; unsigned u; } v; v.f = f;
  unsigned r = v.u + 0x7FFFu + ((v.u >> 16) & 1u);  // round-to-nearest-even
  return (unsigned short)(r >> 16);
}
__device__ __forceinline__ float bf2f(unsigned short h) {
  union { unsigned u; float f; } v; v.u = ((unsigned)h) << 16; return v.f;
}

// ---------------------------------------------------------------------------
// Pack conv weights (L,K,Fin,Fout) f32 -> WMMA B-fragment bf16 layout.
// flat = (((set*16 + w)*NCHUNK + c)*32 + lane)*16 + e
// B element: K-row r = c*32 + (lane<16?0:16) + e ; col = w*16 + (lane&15)
// ---------------------------------------------------------------------------
__global__ __launch_bounds__(256) void pack_weights(
    const float* __restrict__ cw0, const float* __restrict__ cw1,
    const float* __restrict__ cw2, unsigned short* __restrict__ dst) {
  int g = blockIdx.x * 256 + threadIdx.x;
  int e = g & 15;
  int l = (g >> 4) & 31;
  int t = g >> 9;
  int c = t % NCHUNK; t /= NCHUNK;
  int w = t & 15;
  int set = t >> 4;
  if (set >= 6) return;
  int pred = set >> 1, layer = set & 1;
  const float* cw = (pred == 0) ? cw0 : ((pred == 1) ? cw1 : cw2);
  int r  = c * 32 + ((l < 16) ? 0 : 16) + e;
  int k  = r >> 8, fi = r & 255;
  int fo = w * 16 + (l & 15);
  float v = cw[(((layer * KW + k) * F_ + fi) * F_) + fo];
  dst[g] = f2bf_raw(v);
}

// ---------------------------------------------------------------------------
// Produce bf16 gather sources: xbf = bf16(x),
// pxsum = bf16(x + pitch*pc1_w + pc1_b), exsum = bf16(x + energy*ec1_w + ec1_b)
// ---------------------------------------------------------------------------
__global__ __launch_bounds__(256) void make_sums(
    const float* __restrict__ x, const float* __restrict__ pitch,
    const float* __restrict__ energy,
    const float* __restrict__ pc1w, const float* __restrict__ pc1b,
    const float* __restrict__ ec1w, const float* __restrict__ ec1b,
    unsigned short* __restrict__ xbf,
    unsigned short* __restrict__ px, unsigned short* __restrict__ ex) {
  int i = blockIdx.x;            // b*S + s
  int f = threadIdx.x;
  float xv = x[(size_t)i * F_ + f];
  float pv = pitch[i], ev = energy[i];
  xbf[(size_t)i * F_ + f] = f2bf_raw(xv);
  px[(size_t)i * F_ + f]  = f2bf_raw(xv + pv * pc1w[f] + pc1b[f]);
  ex[(size_t)i * F_ + f]  = f2bf_raw(xv + ev * ec1w[f] + ec1b[f]);
}

// ---------------------------------------------------------------------------
// Per-batch duration cumsum + searchsorted + mel_len / mel_mask / dur outputs
// ---------------------------------------------------------------------------
__global__ __launch_bounds__(512) void scan_kernel(
    const int* __restrict__ dur, int* __restrict__ idx, int* __restrict__ mellen,
    float* __restrict__ out_dur, float* __restrict__ out_mellen,
    float* __restrict__ out_melmask) {
  __shared__ int c[S_];
  int b = blockIdx.x, tid = threadIdx.x;
  int d = dur[b * S_ + tid];
  c[tid] = d;
  for (int off = 1; off < S_; off <<= 1) {
    __syncthreads();
    int v = (tid >= off) ? c[tid - off] : 0;
    __syncthreads();
    c[tid] += v;
  }
  __syncthreads();
  int mel = c[S_ - 1];
  if (tid == 0) { mellen[b] = mel; out_mellen[b] = (float)mel; }
  out_dur[b * S_ + tid] = (float)d;
  for (int t = tid; t < T_; t += S_) {
    int lo = 0, hi = S_;                 // upper_bound: first pos with c[pos] > t
    while (lo < hi) { int mid = (lo + hi) >> 1; if (c[mid] <= t) lo = mid + 1; else hi = mid; }
    int id = lo; if (id > S_ - 1) id = S_ - 1;
    idx[b * T_ + t] = id;
    out_melmask[b * T_ + t] = (t < mel) ? 0.0f : 1.0f;
  }
}

// ---------------------------------------------------------------------------
// Fused conv1d(K=3) + bias + ReLU + LayerNorm via v_wmma_f32_16x16x32_bf16.
// Workgroup: 512 threads (16 waves). Tile: 16 time steps x 256 channels.
// A tile staged with global_load_async_to_lds_b128 (one wave-instruction per
// 512B row segment; validity is wave-uniform so padding/invalid segments are
// zero-filled with a uniform ds_store_b128). Output stored as bf16.
// ---------------------------------------------------------------------------
__global__ __launch_bounds__(512) void conv_ln_wmma(
    const unsigned short* __restrict__ in, const unsigned short* __restrict__ Wp,
    const float* __restrict__ bias, const float* __restrict__ gamma,
    const float* __restrict__ beta, unsigned short* __restrict__ out,
    int Tdim, int gather, const int* __restrict__ idx,
    const int* __restrict__ mellen) {
  __shared__ unsigned short Abuf[16 * LDA];
  __shared__ float rowsum[16], rowsq[16], rowmean[16], rowrstd[16];

  const int tid  = threadIdx.x;
  const int lane = tid & 31;
  const int wave = tid >> 5;
  const int b    = blockIdx.y;
  const int t0   = blockIdx.x * 16;

  if (tid < 16) { rowsum[tid] = 0.f; rowsq[tid] = 0.f; }

  // ---- async A-tile stage: wave w stages time row t0+w as 3x512B segments ----
  {
    const int trow = t0 + wave;
    const int ml = gather ? mellen[b] : 0;
    const unsigned abase = (unsigned)(uintptr_t)(&Abuf[0]);
#pragma unroll
    for (int k = 0; k < KW; ++k) {
      int tp = trow + k - 1;             // SAME padding
      bool valid = (tp >= 0) && (tp < Tdim) && (!gather || tp < ml);
      if (valid) {
        const unsigned short* rp;
        if (gather) rp = in + ((size_t)b * S_ + idx[b * Tdim + tp]) * F_;  // length-regulate
        else        rp = in + ((size_t)b * Tdim + tp) * F_;
        unsigned loff = abase + (unsigned)((wave * LDA + k * 256) * 2 + lane * 16);
        asm volatile("global_load_async_to_lds_b128 %0, %1, off"
                     :: "v"(loff),
                        "v"((unsigned long long)(uintptr_t)(rp + lane * 8))
                     : "memory");
      } else {
        uint4 z; z.x = 0; z.y = 0; z.z = 0; z.w = 0;
        *(uint4*)(&Abuf[wave * LDA + k * 256 + lane * 8]) = z;
      }
    }
    asm volatile("s_wait_asynccnt 0x0" ::: "memory");
  }
  __syncthreads();

  union AF { v16bf v; unsigned u[8]; };
  union BF { v16bf v; uint4 q[2]; };
  union CF { v8f  v; float   f[8]; };

  const int m    = lane & 15;
  const int half = lane >> 4;   // 0: lanes 0-15, 1: lanes 16-31
  const int kbA  = half * 8;    // A-fragment K sub-offset per ISA layout

  CF acc;
#pragma unroll
  for (int i = 0; i < 8; ++i) acc.f[i] = 0.f;

  const uint4* wbase = (const uint4*)(Wp + (size_t)wave * NCHUNK * 32 * 16);

  for (int c = 0; c < NCHUNK; ++c) {
    AF a;
#pragma unroll
    for (int v = 0; v < 8; ++v) {
      // 16-bit A 16x32 layout: vgpr v holds K pair {base, base+1}
      int k0 = c * 32 + half * 16 + 2 * v;
      a.u[v] = *(const unsigned*)&Abuf[m * LDA + k0];
    }
    BF bf;
    const uint4* p = wbase + (size_t)(c * 32 + lane) * 2;  // 32B contiguous per lane
    bf.q[0] = p[0];
    bf.q[1] = p[1];
    acc.v = __builtin_amdgcn_wmma_f32_16x16x32_bf16(
        false, a.v, false, bf.v, (short)0, acc.v, false, false);
  }

  // ---- epilogue: bias + ReLU, then LayerNorm over the 256-wide channel axis ----
  const int fo = wave * 16 + m;
  const float bv  = bias[fo];
  const float gv  = gamma[fo];
  const float btv = beta[fo];
  float y[8];
#pragma unroll
  for (int v = 0; v < 8; ++v) {
    float t = acc.f[v] + bv;
    y[v] = t > 0.f ? t : 0.f;
  }
#pragma unroll
  for (int v = 0; v < 8; ++v) {
    float s = y[v], q = y[v] * y[v];
#pragma unroll
    for (int off = 1; off < 16; off <<= 1) {   // reduce over this wave's 16 channels
      s += __shfl_xor(s, off, 32);
      q += __shfl_xor(q, off, 32);
    }
    if (m == v) {                              // one lane per half commits row v(+8)
      int row = v + half * 8;
      atomicAdd(&rowsum[row], s);
      atomicAdd(&rowsq[row], q);
    }
  }
  __syncthreads();
  if (tid < 16) {
    float mean = rowsum[tid] * (1.f / 256.f);
    float var  = rowsq[tid] * (1.f / 256.f) - mean * mean;
    rowmean[tid] = mean;
    rowrstd[tid] = rsqrtf(var + 1e-5f);
  }
  __syncthreads();
#pragma unroll
  for (int v = 0; v < 8; ++v) {
    int row = v + half * 8;
    float o = (y[v] - rowmean[row]) * rowrstd[row] * gv + btv;
    out[((size_t)b * Tdim + t0 + row) * F_ + fo] = f2bf_raw(o);
  }
}

// ---------------------------------------------------------------------------
// Final F->1 projection + mask. One wave per output row. h is bf16.
// ---------------------------------------------------------------------------
__global__ __launch_bounds__(256) void linear_mask(
    const unsigned short* __restrict__ h, const float* __restrict__ lw,
    const float* __restrict__ lb, float* __restrict__ o, int Tdim, int use_mel,
    const unsigned char* __restrict__ smask, const int* __restrict__ mellen) {
  int lane = threadIdx.x & 31;
  int wv   = threadIdx.x >> 5;
  long r = (long)blockIdx.x * 8 + wv;
  int b = (int)(r / Tdim), t = (int)(r % Tdim);
  const unsigned short* hp = h + (size_t)r * F_;
  float s = 0.f;
  for (int f = lane; f < F_; f += 32) s += bf2f(hp[f]) * lw[f];
  for (int off = 1; off < 32; off <<= 1) s += __shfl_xor(s, off, 32);
  if (lane == 0) {
    s += lb[0];
    bool msk = use_mel ? (t >= mellen[b]) : (smask[(size_t)b * Tdim + t] != 0);
    o[r] = msk ? 0.f : s;
  }
}

// ---------------------------------------------------------------------------
// out = gather(x)*valid + pitch_pred*pc2_w + pc2_b + energy_pred*ec2_w + ec2_b
// (x kept f32 here: the residual term stays full precision)
// ---------------------------------------------------------------------------
__global__ __launch_bounds__(256) void final_out(
    const float* __restrict__ x, const int* __restrict__ idx,
    const int* __restrict__ mellen,
    const float* __restrict__ pp, const float* __restrict__ ep,
    const float* __restrict__ pc2w, const float* __restrict__ pc2b,
    const float* __restrict__ ec2w, const float* __restrict__ ec2b,
    float* __restrict__ out) {
  int i = blockIdx.x;           // b*T + t
  int b = i >> 11;              // T_ = 2048
  int t = i & (T_ - 1);
  int f = threadIdx.x;
  float xe = 0.f;
  if (t < mellen[b]) xe = x[((size_t)b * S_ + idx[i]) * F_ + f];
  float p = pp[i], e = ep[i];
  out[(size_t)i * F_ + f] = xe + p * pc2w[f] + pc2b[f] + e * ec2w[f] + ec2b[f];
}

// ---------------------------------------------------------------------------
extern "C" void kernel_launch(void* const* d_in, const int* in_sizes, int n_in,
                              void* d_out, int out_size, void* d_ws, size_t ws_size,
                              hipStream_t stream) {
  (void)in_sizes; (void)n_in; (void)out_size; (void)ws_size;
  // inputs (setup_inputs order)
  const float* x            = (const float*)d_in[0];
  const unsigned char* smsk = (const unsigned char*)d_in[1];
  const float* src_pitch    = (const float*)d_in[3];
  const float* src_energy   = (const float*)d_in[4];
  const int*   src_duration = (const int*)d_in[5];
  const float* dur_cw = (const float*)d_in[7];
  const float* dur_cb = (const float*)d_in[8];
  const float* dur_lng= (const float*)d_in[9];
  const float* dur_lnb= (const float*)d_in[10];
  const float* dur_lw = (const float*)d_in[11];
  const float* dur_lb = (const float*)d_in[12];
  const float* pit_cw = (const float*)d_in[13];
  const float* pit_cb = (const float*)d_in[14];
  const float* pit_lng= (const float*)d_in[15];
  const float* pit_lnb= (const float*)d_in[16];
  const float* pit_lw = (const float*)d_in[17];
  const float* pit_lb = (const float*)d_in[18];
  const float* ene_cw = (const float*)d_in[19];
  const float* ene_cb = (const float*)d_in[20];
  const float* ene_lng= (const float*)d_in[21];
  const float* ene_lnb= (const float*)d_in[22];
  const float* ene_lw = (const float*)d_in[23];
  const float* ene_lb = (const float*)d_in[24];
  const float* pc1w = (const float*)d_in[25];
  const float* pc1b = (const float*)d_in[26];
  const float* pc2w = (const float*)d_in[27];
  const float* pc2b = (const float*)d_in[28];
  const float* ec1w = (const float*)d_in[29];
  const float* ec1b = (const float*)d_in[30];
  const float* ec2w = (const float*)d_in[31];
  const float* ec2b = (const float*)d_in[32];

  // output layout (floats, reference tuple order)
  float* outf = (float*)d_out;
  const size_t PP_OFF   = (size_t)B_ * T_ * H_;                // pitch_pred (B,T)
  const size_t EP_OFF   = PP_OFF + (size_t)B_ * T_;            // energy_pred
  const size_t LOGD_OFF = EP_OFF + (size_t)B_ * T_;            // log_d (B,S)
  const size_t DUR_OFF  = LOGD_OFF + (size_t)B_ * S_;          // dur (B,S)
  const size_t MEL_OFF  = DUR_OFF + (size_t)B_ * S_;           // mel_len (B)
  const size_t MASK_OFF = MEL_OFF + (size_t)B_;                // mel_mask (B,T)

  // workspace layout (bytes, all 256B aligned)
  char* ws = (char*)d_ws;
  unsigned short* WP    = (unsigned short*)(ws + 0);           // 6 * 384KB packed bf16
  int*            idx   = (int*)(ws + 2359296);                // B*T
  int*            mellen= (int*)(ws + 2621440);                // B
  unsigned short* xbf   = (unsigned short*)(ws + 2621696);     // B*S*F bf16
  unsigned short* pxsum = (unsigned short*)(ws + 11010304);    // B*S*F bf16
  unsigned short* exsum = (unsigned short*)(ws + 19398912);    // B*S*F bf16
  unsigned short* tmp0  = (unsigned short*)(ws + 27787520);    // B*T*F bf16
  unsigned short* tmp1  = (unsigned short*)(ws + 61341952);    // B*T*F bf16

  // 1) pack all 6 conv-weight sets into the WMMA B-fragment layout
  pack_weights<<<(6 * SETSZ) / 256, 256, 0, stream>>>(dur_cw, pit_cw, ene_cw, WP);

  // 2) bf16 gather sources: x, x + pitch emb, x + energy emb (pre-regulation)
  make_sums<<<B_ * S_, 256, 0, stream>>>(x, src_pitch, src_energy,
                                         pc1w, pc1b, ec1w, ec1b, xbf, pxsum, exsum);

  // 3) duration cumsum / searchsorted / mel_len / mel_mask / dur outputs
  scan_kernel<<<B_, 512, 0, stream>>>(src_duration, idx, mellen,
                                      outf + DUR_OFF, outf + MEL_OFF, outf + MASK_OFF);

  // 4) duration tower (on x, T=S)
  {
    dim3 g(S_ / 16, B_);
    conv_ln_wmma<<<g, 512, 0, stream>>>(xbf,  WP + 0 * SETSZ, dur_cb,      dur_lng,      dur_lnb,      tmp0, S_, 0, idx, mellen);
    conv_ln_wmma<<<g, 512, 0, stream>>>(tmp0, WP + 1 * SETSZ, dur_cb + F_, dur_lng + F_, dur_lnb + F_, tmp1, S_, 0, idx, mellen);
    linear_mask<<<(B_ * S_) / 8, 256, 0, stream>>>(tmp1, dur_lw, dur_lb,
                                                   outf + LOGD_OFF, S_, 0, smsk, mellen);
  }

  // 5) pitch tower (layer 0 gathers pe = length_regulate(x + pitch_conv) on the fly)
  {
    dim3 g(T_ / 16, B_);
    conv_ln_wmma<<<g, 512, 0, stream>>>(pxsum, WP + 2 * SETSZ, pit_cb,      pit_lng,      pit_lnb,      tmp0, T_, 1, idx, mellen);
    conv_ln_wmma<<<g, 512, 0, stream>>>(tmp0,  WP + 3 * SETSZ, pit_cb + F_, pit_lng + F_, pit_lnb + F_, tmp1, T_, 0, idx, mellen);
    linear_mask<<<(B_ * T_) / 8, 256, 0, stream>>>(tmp1, pit_lw, pit_lb,
                                                   outf + PP_OFF, T_, 1, smsk, mellen);
  }

  // 6) energy tower
  {
    dim3 g(T_ / 16, B_);
    conv_ln_wmma<<<g, 512, 0, stream>>>(exsum, WP + 4 * SETSZ, ene_cb,      ene_lng,      ene_lnb,      tmp0, T_, 1, idx, mellen);
    conv_ln_wmma<<<g, 512, 0, stream>>>(tmp0,  WP + 5 * SETSZ, ene_cb + F_, ene_lng + F_, ene_lnb + F_, tmp1, T_, 0, idx, mellen);
    linear_mask<<<(B_ * T_) / 8, 256, 0, stream>>>(tmp1, ene_lw, ene_lb,
                                                   outf + EP_OFF, T_, 1, smsk, mellen);
  }

  // 7) out = xe + pitch_emb + energy_emb (xe re-gathered on the fly)
  final_out<<<B_ * T_, 256, 0, stream>>>(x, idx, mellen,
                                         outf + PP_OFF, outf + EP_OFF,
                                         pc2w, pc2b, ec2w, ec2b, outf);
}
